// HeteroGVPConv_85718957293691
// MI455X (gfx1250) — matrix operand order repaired and checked by reading, not compiled, not run on hardware
//
#include <hip/hip_runtime.h>

typedef __attribute__((ext_vector_type(16))) _Float16 v16h;
typedef __attribute__((ext_vector_type(8)))  float    v8f;

#define N_NODES 50000
#define N_EDGES 800000
#define S_DIM   128
#define V_DIM   16
#define DVH_M   17     // V+1 hidden vectors (message GVP)
#define KM      161    // 128 (s) + 16 (rbf) + 17 (sh)
#define KM_PAD  192    // 6 K-steps of 32
#define DVH_U   32     // 2V hidden vectors (update GVP)
#define KU      288    // 128 (s) + 128 (agg_s) + 32 (sh) = 9 K-steps of 32
#define EPSV    1e-8f

// packed B-fragment counts (in v16h fragments of 32 lanes * 16 halves)
#define FRAG_WFM (6*8*32)   // 6 ksteps x 8 ntiles
#define FRAG_WGM (4*1*32)   // 4 ksteps x 1 ntile
#define FRAG_WFU (9*8*32)
#define FRAG_WGU (4*1*32)

__device__ __forceinline__ float sigmoidf(float x) { return 1.f / (1.f + __expf(-x)); }

union H4 { _Float16 h[4]; uint2 u2; };

// ---------------------------------------------------------------------------
// Kernel 0a: zero the aggregation workspace (agg_s [N,128] ++ agg_v [N,16,3])
// ---------------------------------------------------------------------------
__global__ void gvp_zero_ws(float* __restrict__ p, int n) {
    int i = blockIdx.x * blockDim.x + threadIdx.x;
    if (i < n) p[i] = 0.f;
}

// ---------------------------------------------------------------------------
// Kernel 0b: pre-pack a weight matrix W[Krows x Ncols] (row-major, K-major)
// into WMMA B-fragment layout: out[((k*ntiles+n)*32 + lane)*16 + h], where
// K = k*32 + (h/8)*16 + (lane/16)*8 + h%8, N = n*16 + (lane%16). Rows K >= Krows -> 0.
// ---------------------------------------------------------------------------
__global__ void gvp_pack_b(const float* __restrict__ W, _Float16* __restrict__ out,
                           int Krows, int Ncols, int ntiles, int total) {
    int idx = blockIdx.x * blockDim.x + threadIdx.x;
    if (idx >= total) return;
    int h    = idx & 15;
    int lane = (idx >> 4) & 31;
    int kn   = idx >> 9;
    int n    = kn % ntiles;
    int k    = kn / ntiles;
    int K = k*32 + (h >> 3)*16 + (lane >> 4)*8 + (h & 7);
    int N = n*16 + (lane & 15);
    float val = (K < Krows) ? W[K*Ncols + N] : 0.f;
    out[idx] = (_Float16)val;
}

// ---------------------------------------------------------------------------
// Kernel 1: edge messages. 4 waves/block, 16 edges per wave (one WMMA M-tile)
// ---------------------------------------------------------------------------
__global__ __launch_bounds__(128) void gvp_edge_kernel(
    const float* __restrict__ s, const float* __restrict__ v, const float* __restrict__ x,
    const float* __restrict__ Wh, const float* __restrict__ Wu,
    const v16h* __restrict__ BWf, const float* __restrict__ bf,
    const v16h* __restrict__ BWg, const float* __restrict__ bg,
    const int* __restrict__ ei,
    float* __restrict__ agg_s, float* __restrict__ agg_v)
{
    __shared__ __align__(16) _Float16 sA[4][16][KM_PAD];   // A-tile staging
    __shared__ __align__(16) _Float16 sF[4][16][S_DIM];    // feats f16
    __shared__ float    sG[4][16][V_DIM];                  // gate values
    __shared__ int      sDst[4][16];

    const int wave = threadIdx.x >> 5;
    const int lane = threadIdx.x & 31;
    const int li   = lane & 15;
    const int half = lane >> 4;
    const int eg   = blockIdx.x * 64 + wave * 16 + li;
    const bool valid = eg < N_EDGES;

    int src = 0, dst = 0;
    if (valid) { src = ei[eg]; dst = ei[N_EDGES + eg]; }
    if (half == 0) sDst[wave][li] = valid ? dst : -1;

    // ---- geometry ----
    float dx0 = x[dst*3+0] - x[src*3+0];
    float dx1 = x[dst*3+1] - x[src*3+1];
    float dx2 = x[dst*3+2] - x[src*3+2];
    float dist = sqrtf(fmaxf(dx0*dx0 + dx1*dx1 + dx2*dx2, EPSV));
    float inv  = 1.f / dist;
    float u0 = dx0*inv, u1 = dx1*inv, u2 = dx2*inv;

    // ---- stage s[src] (cols 0..127), packed 4-wide f16 stores ----
    {
        const float4* sp = (const float4*)(s + (size_t)src * S_DIM);
        #pragma unroll
        for (int j = 0; j < 16; ++j) {
            float4 q = sp[half*16 + j];
            H4 pk;
            pk.h[0] = (_Float16)q.x; pk.h[1] = (_Float16)q.y;
            pk.h[2] = (_Float16)q.z; pk.h[3] = (_Float16)q.w;
            *(uint2*)&sA[wave][li][half*64 + j*4] = pk.u2;
        }
    }
    // ---- rbf(dist) (cols 128..143): mu_i = i*20/15, sigma = 1.25 ----
    #pragma unroll
    for (int j = 0; j < 2; ++j) {
        H4 pk;
        #pragma unroll
        for (int q = 0; q < 4; ++q) {
            int i = half*8 + j*4 + q;
            float t = (dist - (20.f/15.f)*(float)i) * (1.f/1.25f);
            pk.h[q] = (_Float16)__expf(-t*t);
        }
        *(uint2*)&sA[wave][li][128 + half*8 + j*4] = pk.u2;
    }
    // ---- zero cols 160..191 (sh[16] at col 160 is overwritten below) ----
    {
        H4 z; z.h[0] = (_Float16)0.f; z.h[1] = (_Float16)0.f;
        z.h[2] = (_Float16)0.f; z.h[3] = (_Float16)0.f;
        #pragma unroll
        for (int j = 0; j < 4; ++j)
            *(uint2*)&sA[wave][li][160 + half*16 + j*4] = z.u2;
    }

    // ---- vector path: Vh = v_in^T Wh (17x17), sh (cols 144..160), Vu ----
    float Vu[V_DIM][3];
    #pragma unroll
    for (int o = 0; o < V_DIM; ++o) { Vu[o][0]=0.f; Vu[o][1]=0.f; Vu[o][2]=0.f; }
    {
        float vin[DVH_M][3];
        #pragma unroll
        for (int o = 0; o < V_DIM; ++o) {
            vin[o][0] = v[((size_t)src*V_DIM + o)*3 + 0];
            vin[o][1] = v[((size_t)src*V_DIM + o)*3 + 1];
            vin[o][2] = v[((size_t)src*V_DIM + o)*3 + 2];
        }
        vin[16][0] = u0; vin[16][1] = u1; vin[16][2] = u2;
        int h0 = (half == 0) ? 0 : 9;
        int h1 = (half == 0) ? 9 : 17;
        for (int h = h0; h < h1; ++h) {
            float a0 = 0.f, a1 = 0.f, a2 = 0.f;
            #pragma unroll
            for (int w = 0; w < DVH_M; ++w) {
                float wv = Wh[w*DVH_M + h];
                a0 += vin[w][0]*wv; a1 += vin[w][1]*wv; a2 += vin[w][2]*wv;
            }
            sA[wave][li][144 + h] = (_Float16)sqrtf(fmaxf(a0*a0 + a1*a1 + a2*a2, EPSV));
            #pragma unroll
            for (int o = 0; o < V_DIM; ++o) {
                float wu = Wu[h*V_DIM + o];
                Vu[o][0] += a0*wu; Vu[o][1] += a1*wu; Vu[o][2] += a2*wu;
            }
        }
    }
    #pragma unroll
    for (int o = 0; o < V_DIM; ++o) {
        Vu[o][0] += __shfl_xor(Vu[o][0], 16, 32);
        Vu[o][1] += __shfl_xor(Vu[o][1], 16, 32);
        Vu[o][2] += __shfl_xor(Vu[o][2], 16, 32);
    }
    __syncthreads();

    // ---- big GEMM: feats = silu([s|rbf|sh] @ Wf + bf), M=16 N=128 K=192 ----
    v8f acc[8];
    #pragma unroll
    for (int n = 0; n < 8; ++n) { v8f z = {}; acc[n] = z; }

    #pragma unroll 2
    for (int k = 0; k < 6; ++k) {
        v16h a;  // A 16x32 f16: M=lane%16, K = k*32 + (h/8)*16 + half*8 + h%8
        const _Float16* ap = &sA[wave][li][k*32 + half*8];
        #pragma unroll
        for (int h = 0; h < 8; ++h) { a[h] = ap[h]; a[8+h] = ap[16+h]; }
        #pragma unroll
        for (int n = 0; n < 8; ++n) {
            v16h b = BWf[(k*8 + n)*32 + lane];   // pre-packed fragment
            acc[n] = __builtin_amdgcn_wmma_f32_16x16x32_f16(
                false, a, false, b, (short)0, acc[n], false, false);
        }
    }

    // ---- bias + SiLU; stage feats f16 for gating GEMM ----
    #pragma unroll
    for (int n = 0; n < 8; ++n) {
        float bias = bf[n*16 + li];
        #pragma unroll
        for (int r = 0; r < 8; ++r) {
            float z = acc[n][r] + bias;
            float f = z * sigmoidf(z);
            acc[n][r] = f;
            sF[wave][r + half*8][n*16 + li] = (_Float16)f;
        }
    }
    __syncthreads();

    // ---- gate = sigmoid(feats @ Wg + bg), K=128 ----
    v8f accg; { v8f z = {}; accg = z; }
    #pragma unroll
    for (int k = 0; k < 4; ++k) {
        v16h a;
        const _Float16* ap = &sF[wave][li][k*32 + half*8];
        #pragma unroll
        for (int h = 0; h < 8; ++h) { a[h] = ap[h]; a[8+h] = ap[16+h]; }
        v16h b = BWg[k*32 + lane];
        accg = __builtin_amdgcn_wmma_f32_16x16x32_f16(
            false, a, false, b, (short)0, accg, false, false);
    }
    {
        float bgc = bg[li];
        #pragma unroll
        for (int r = 0; r < 8; ++r)
            sG[wave][r + half*8][li] = sigmoidf(accg[r] + bgc);
    }
    __syncthreads();

    // ---- scatter scalar messages (pre-scaled by 1/MSG_NORM = 0.1) ----
    #pragma unroll
    for (int r = 0; r < 8; ++r) {
        int d = sDst[wave][r + half*8];
        if (d >= 0) {
            #pragma unroll
            for (int n = 0; n < 8; ++n)
                atomicAdd(&agg_s[(size_t)d*S_DIM + n*16 + li], acc[n][r] * 0.1f);
        }
    }
    // ---- scatter vector messages: mv = gate * Vu ----
    if (valid) {
        #pragma unroll
        for (int j = 0; j < 8; ++j) {
            int o = half*8 + j;
            float g = sG[wave][li][o] * 0.1f;
            atomicAdd(&agg_v[((size_t)dst*V_DIM + o)*3 + 0], g * Vu[o][0]);
            atomicAdd(&agg_v[((size_t)dst*V_DIM + o)*3 + 1], g * Vu[o][1]);
            atomicAdd(&agg_v[((size_t)dst*V_DIM + o)*3 + 2], g * Vu[o][2]);
        }
    }
}

// ---------------------------------------------------------------------------
// Kernel 2: update GVP + residual + GVPLayerNorm. 16 nodes per wave.
// ---------------------------------------------------------------------------
__global__ __launch_bounds__(128) void gvp_node_kernel(
    const float* __restrict__ s, const float* __restrict__ v,
    const float* __restrict__ agg_s, const float* __restrict__ agg_v,
    const float* __restrict__ Wh, const float* __restrict__ Wu,
    const v16h* __restrict__ BWf, const float* __restrict__ bf,
    const v16h* __restrict__ BWg, const float* __restrict__ bg,
    const float* __restrict__ ln_g, const float* __restrict__ ln_b,
    float* __restrict__ out_s, float* __restrict__ out_v)
{
    __shared__ __align__(16) _Float16 sA[4][16][KU];
    __shared__ __align__(16) _Float16 sF[4][16][S_DIM];
    __shared__ float    sG[4][16][V_DIM];

    const int wave = threadIdx.x >> 5;
    const int lane = threadIdx.x & 31;
    const int li   = lane & 15;
    const int half = lane >> 4;
    const int tileBase = blockIdx.x * 64 + wave * 16;
    const int ni = tileBase + li;
    const bool valid = ni < N_NODES;
    const int nid = valid ? ni : 0;

    // ---- stage [s | agg_s] (cols 0..255), packed stores ----
    {
        const float4* sp = (const float4*)(s     + (size_t)nid * S_DIM);
        const float4* ap = (const float4*)(agg_s + (size_t)nid * S_DIM);
        #pragma unroll
        for (int j = 0; j < 16; ++j) {
            float4 q = sp[half*16 + j];
            H4 pk;
            pk.h[0] = (_Float16)q.x; pk.h[1] = (_Float16)q.y;
            pk.h[2] = (_Float16)q.z; pk.h[3] = (_Float16)q.w;
            *(uint2*)&sA[wave][li][half*64 + j*4] = pk.u2;
            float4 q2 = ap[half*16 + j];
            H4 pk2;
            pk2.h[0] = (_Float16)q2.x; pk2.h[1] = (_Float16)q2.y;
            pk2.h[2] = (_Float16)q2.z; pk2.h[3] = (_Float16)q2.w;
            *(uint2*)&sA[wave][li][128 + half*64 + j*4] = pk2.u2;
        }
    }

    // ---- vector path: v_cat = [v | agg_v] (32x3), Vh (32x32), sh, Vu ----
    float Vu[V_DIM][3];
    #pragma unroll
    for (int o = 0; o < V_DIM; ++o) { Vu[o][0]=0.f; Vu[o][1]=0.f; Vu[o][2]=0.f; }
    {
        float vin[DVH_U][3];
        #pragma unroll
        for (int o = 0; o < V_DIM; ++o) {
            vin[o][0]    = v[((size_t)nid*V_DIM + o)*3 + 0];
            vin[o][1]    = v[((size_t)nid*V_DIM + o)*3 + 1];
            vin[o][2]    = v[((size_t)nid*V_DIM + o)*3 + 2];
            vin[16+o][0] = agg_v[((size_t)nid*V_DIM + o)*3 + 0];
            vin[16+o][1] = agg_v[((size_t)nid*V_DIM + o)*3 + 1];
            vin[16+o][2] = agg_v[((size_t)nid*V_DIM + o)*3 + 2];
        }
        for (int h = half*16; h < half*16 + 16; ++h) {
            float a0 = 0.f, a1 = 0.f, a2 = 0.f;
            #pragma unroll
            for (int w = 0; w < DVH_U; ++w) {
                float wv = Wh[w*DVH_U + h];
                a0 += vin[w][0]*wv; a1 += vin[w][1]*wv; a2 += vin[w][2]*wv;
            }
            sA[wave][li][256 + h] = (_Float16)sqrtf(fmaxf(a0*a0 + a1*a1 + a2*a2, EPSV));
            #pragma unroll
            for (int o = 0; o < V_DIM; ++o) {
                float wu = Wu[h*V_DIM + o];
                Vu[o][0] += a0*wu; Vu[o][1] += a1*wu; Vu[o][2] += a2*wu;
            }
        }
    }
    #pragma unroll
    for (int o = 0; o < V_DIM; ++o) {
        Vu[o][0] += __shfl_xor(Vu[o][0], 16, 32);
        Vu[o][1] += __shfl_xor(Vu[o][1], 16, 32);
        Vu[o][2] += __shfl_xor(Vu[o][2], 16, 32);
    }
    __syncthreads();

    // ---- update GEMM: us = silu([s|agg_s|sh] @ Wf_u + bf_u), K=288 ----
    v8f acc[8];
    #pragma unroll
    for (int n = 0; n < 8; ++n) { v8f z = {}; acc[n] = z; }

    #pragma unroll 2
    for (int k = 0; k < 9; ++k) {
        v16h a;
        const _Float16* ap = &sA[wave][li][k*32 + half*8];
        #pragma unroll
        for (int h = 0; h < 8; ++h) { a[h] = ap[h]; a[8+h] = ap[16+h]; }
        #pragma unroll
        for (int n = 0; n < 8; ++n) {
            v16h b = BWf[(k*8 + n)*32 + lane];
            acc[n] = __builtin_amdgcn_wmma_f32_16x16x32_f16(
                false, a, false, b, (short)0, acc[n], false, false);
        }
    }
    #pragma unroll
    for (int n = 0; n < 8; ++n) {
        float bias = bf[n*16 + li];
        #pragma unroll
        for (int r = 0; r < 8; ++r) {
            float z = acc[n][r] + bias;
            float f = z * sigmoidf(z);
            acc[n][r] = f;
            sF[wave][r + half*8][n*16 + li] = (_Float16)f;
        }
    }
    __syncthreads();

    // ---- gate GEMM ----
    v8f accg; { v8f z = {}; accg = z; }
    #pragma unroll
    for (int k = 0; k < 4; ++k) {
        v16h a;
        const _Float16* ap = &sF[wave][li][k*32 + half*8];
        #pragma unroll
        for (int h = 0; h < 8; ++h) { a[h] = ap[h]; a[8+h] = ap[16+h]; }
        v16h b = BWg[k*32 + lane];
        accg = __builtin_amdgcn_wmma_f32_16x16x32_f16(
            false, a, false, b, (short)0, accg, false, false);
    }
    {
        float bgc = bg[li];
        #pragma unroll
        for (int r = 0; r < 8; ++r)
            sG[wave][r + half*8][li] = sigmoidf(accg[r] + bgc);
    }
    __syncthreads();

    // ---- scalar residual + LayerNorm over 128 features per node ----
    float sum1[8], sum2[8];
    #pragma unroll
    for (int r = 0; r < 8; ++r) { sum1[r] = 0.f; sum2[r] = 0.f; }
    #pragma unroll
    for (int n = 0; n < 8; ++n) {
        #pragma unroll
        for (int r = 0; r < 8; ++r) {
            int node = tileBase + r + half*8;
            int nn = (node < N_NODES) ? node : 0;
            float val = s[(size_t)nn*S_DIM + n*16 + li] + acc[n][r];
            acc[n][r] = val;
            sum1[r] += val;
            sum2[r] += val*val;
        }
    }
    #pragma unroll
    for (int m = 1; m < 16; m <<= 1) {
        #pragma unroll
        for (int r = 0; r < 8; ++r) {
            sum1[r] += __shfl_xor(sum1[r], m, 32);
            sum2[r] += __shfl_xor(sum2[r], m, 32);
        }
    }
    #pragma unroll
    for (int n = 0; n < 8; ++n) {
        float lg = ln_g[n*16 + li], lb = ln_b[n*16 + li];
        #pragma unroll
        for (int r = 0; r < 8; ++r) {
            int node = tileBase + r + half*8;
            if (node < N_NODES) {
                float mu  = sum1[r] * (1.f/128.f);
                float var = sum2[r] * (1.f/128.f) - mu*mu;
                out_s[(size_t)node*S_DIM + n*16 + li] =
                    (acc[n][r] - mu) * rsqrtf(var + 1e-5f) * lg + lb;
            }
        }
    }

    // ---- vector residual + vector norm (per node, over 16 vectors) ----
    float vres[8][3];
    float part = 0.f;
    #pragma unroll
    for (int j = 0; j < 8; ++j) {
        int o = half*8 + j;
        float g  = sG[wave][li][o];
        float r0 = v[((size_t)nid*V_DIM + o)*3 + 0] + g * Vu[o][0];
        float r1 = v[((size_t)nid*V_DIM + o)*3 + 1] + g * Vu[o][1];
        float r2 = v[((size_t)nid*V_DIM + o)*3 + 2] + g * Vu[o][2];
        vres[j][0] = r0; vres[j][1] = r1; vres[j][2] = r2;
        part += fmaxf(r0*r0 + r1*r1 + r2*r2, EPSV);
    }
    part += __shfl_xor(part, 16, 32);
    float invn = 1.f / (sqrtf(part * (1.f/16.f) + 1e-5f) + 1e-5f);
    if (valid) {
        #pragma unroll
        for (int j = 0; j < 8; ++j) {
            int o = half*8 + j;
            out_v[((size_t)ni*V_DIM + o)*3 + 0] = vres[j][0] * invn;
            out_v[((size_t)ni*V_DIM + o)*3 + 1] = vres[j][1] * invn;
            out_v[((size_t)ni*V_DIM + o)*3 + 2] = vres[j][2] * invn;
        }
    }
}

// ---------------------------------------------------------------------------
extern "C" void kernel_launch(void* const* d_in, const int* in_sizes, int n_in,
                              void* d_out, int out_size, void* d_ws, size_t ws_size,
                              hipStream_t stream) {
    (void)in_sizes; (void)n_in; (void)out_size; (void)ws_size;
    const float* s    = (const float*)d_in[0];
    const float* v    = (const float*)d_in[1];
    const float* x    = (const float*)d_in[2];
    const float* Wh_m = (const float*)d_in[3];
    const float* Wu_m = (const float*)d_in[4];
    const float* Wf_m = (const float*)d_in[5];
    const float* bf_m = (const float*)d_in[6];
    const float* Wg_m = (const float*)d_in[7];
    const float* bg_m = (const float*)d_in[8];
    const float* Wh_u = (const float*)d_in[9];
    const float* Wu_u = (const float*)d_in[10];
    const float* Wf_u = (const float*)d_in[11];
    const float* bf_u = (const float*)d_in[12];
    const float* Wg_u = (const float*)d_in[13];
    const float* bg_u = (const float*)d_in[14];
    const float* ln_g = (const float*)d_in[15];
    const float* ln_b = (const float*)d_in[16];
    const int*   ei   = (const int*)d_in[17];

    float* agg_s = (float*)d_ws;                       // [N,128]
    float* agg_v = agg_s + (size_t)N_NODES * S_DIM;    // [N,16,3]
    _Float16* packBase = (_Float16*)(agg_v + (size_t)N_NODES * V_DIM * 3);
    _Float16* pWfM = packBase;                         // 6*8*512 halves
    _Float16* pWgM = pWfM + (size_t)FRAG_WFM * 16;     // 4*512
    _Float16* pWfU = pWgM + (size_t)FRAG_WGM * 16;     // 9*8*512
    _Float16* pWgU = pWfU + (size_t)FRAG_WFU * 16;     // 4*512

    float* out_s = (float*)d_out;                      // [N,128]
    float* out_v = out_s + (size_t)N_NODES * S_DIM;    // [N,16,3]

    // prep: zero aggregation + pack weights into WMMA B-fragment layout
    const int zn = N_NODES * (S_DIM + V_DIM * 3);
    gvp_zero_ws<<<(zn + 255) / 256, 256, 0, stream>>>(agg_s, zn);
    gvp_pack_b<<<(FRAG_WFM*16 + 255)/256, 256, 0, stream>>>(Wf_m, pWfM, KM,    S_DIM, 8, FRAG_WFM*16);
    gvp_pack_b<<<(FRAG_WGM*16 + 255)/256, 256, 0, stream>>>(Wg_m, pWgM, S_DIM, V_DIM, 1, FRAG_WGM*16);
    gvp_pack_b<<<(FRAG_WFU*16 + 255)/256, 256, 0, stream>>>(Wf_u, pWfU, KU,    S_DIM, 8, FRAG_WFU*16);
    gvp_pack_b<<<(FRAG_WGU*16 + 255)/256, 256, 0, stream>>>(Wg_u, pWgU, S_DIM, V_DIM, 1, FRAG_WGU*16);

    gvp_edge_kernel<<<(N_EDGES + 63) / 64, 128, 0, stream>>>(
        s, v, x, Wh_m, Wu_m, (const v16h*)pWfM, bf_m, (const v16h*)pWgM, bg_m,
        ei, agg_s, agg_v);

    gvp_node_kernel<<<(N_NODES + 63) / 64, 128, 0, stream>>>(
        s, v, agg_s, agg_v, Wh_u, Wu_u, (const v16h*)pWfU, bf_u, (const v16h*)pWgU, bg_u,
        ln_g, ln_b, out_s, out_v);
}
